// attention_node_to_node_19258633355663
// MI455X (gfx1250) — matrix operand order repaired and checked by reading, not compiled, not run on hardware
//
#include <hip/hip_runtime.h>
#include <cstdint>

#define D_MODEL 512
#define DH      64
#define BATCH   8
#define SEQ     4096

typedef __attribute__((ext_vector_type(16))) __bf16 v16bf;
typedef __attribute__((ext_vector_type(8)))  __bf16 v8bf;
typedef __attribute__((ext_vector_type(8)))  float  v8f;

// ---- WMMA fragment helpers (CDNA5 wave32 layouts, CDNA5 ISA) ----

// A-frag 16x32 bf16: lane (r = lane&15, h = lane>>4) holds row r,
// elements 0..7 = K (h*8 .. h*8+7), elements 8..15 = K (16+h*8 .. 23+h*8).
__device__ __forceinline__ v16bf load_a_bf16(const __bf16* rowp, int h) {
    v8bf lo = *(const v8bf*)(rowp + h * 8);
    v8bf hi = *(const v8bf*)(rowp + 16 + h * 8);
    return __builtin_shufflevector(lo, hi, 0, 1, 2, 3, 4, 5, 6, 7,
                                           8, 9, 10, 11, 12, 13, 14, 15);
}

// A-frag built from fp32 source with on-the-fly bf16 conversion.
__device__ __forceinline__ v16bf load_a_f32(const float* rowp, int h) {
    const float* p = rowp + h * 8;
    v16bf a;
#pragma unroll
    for (int i = 0; i < 8; ++i) a[i] = (__bf16)p[i];
#pragma unroll
    for (int i = 0; i < 8; ++i) a[8 + i] = (__bf16)p[16 + i];
    return a;
}

// B-frag 32x16 bf16: lane holds column n = lane&15; elements 0..7 = K (h*8 .. h*8+7), elements 8..15 = K (16+h*8 .. 23+h*8).
__device__ __forceinline__ v16bf load_b_bf16(const __bf16* colp, int h) {
    v8bf lo = *(const v8bf*)(colp + h * 8);
    v8bf hi = *(const v8bf*)(colp + 16 + h * 8);
    return __builtin_shufflevector(lo, hi, 0, 1, 2, 3, 4, 5, 6, 7,
                                           8, 9, 10, 11, 12, 13, 14, 15);
}

__device__ __forceinline__ v8f wmma_bf16(v16bf a, v16bf b, v8f c) {
    return __builtin_amdgcn_wmma_f32_16x16x32_bf16(false, a, false, b,
                                                   (short)0, c, false, false);
}

// ---- Async global->LDS b128 copy (CDNA5 GLOBAL_LOAD_ASYNC_TO_LDS_B128) ----
// Builtin exists on this toolchain with param0 = v4i __device__* (global src).
// SFINAE cascade: try (gsrc, ldsdst), then (ldsdst, gsrc), else sync fallback,
// so any address-space/arity mismatch degrades gracefully instead of erroring.
typedef int v4i __attribute__((vector_size(16)));
typedef __attribute__((address_space(1))) v4i* as1_v4i;
typedef __attribute__((address_space(3))) v4i* as3_v4i;

struct rank0 {}; struct rank1 : rank0 {}; struct rank2 : rank1 {};

#if __has_builtin(__builtin_amdgcn_global_load_async_to_lds_b128)
template <typename G, typename L>
__device__ __forceinline__ auto g2l_impl(G g, L l, __bf16*, const __bf16*, rank2)
    -> decltype(__builtin_amdgcn_global_load_async_to_lds_b128(g, l, 0, 0), void()) {
    __builtin_amdgcn_global_load_async_to_lds_b128(g, l, 0, 0);
}
template <typename G, typename L>
__device__ __forceinline__ auto g2l_impl(G g, L l, __bf16*, const __bf16*, rank1)
    -> decltype(__builtin_amdgcn_global_load_async_to_lds_b128(l, g, 0, 0), void()) {
    __builtin_amdgcn_global_load_async_to_lds_b128(l, g, 0, 0);
}
#endif
template <typename G, typename L>
__device__ __forceinline__ void g2l_impl(G, L, __bf16* lds, const __bf16* g, rank0) {
    *(v8bf*)lds = *(const v8bf*)g;      // synchronous fallback
}

__device__ __forceinline__ void copy_b128_g2l(__bf16* lds, const __bf16* g) {
    g2l_impl((as1_v4i)(uintptr_t)g, (as3_v4i)(uintptr_t)lds, lds, g, rank2{});
}

__device__ __forceinline__ void wait_async0() {
#if __has_builtin(__builtin_amdgcn_s_wait_asynccnt)
    __builtin_amdgcn_s_wait_asynccnt(0);
#else
    asm volatile("s_wait_asynccnt 0" ::: "memory");
#endif
}

// ---- Kernel 0: transpose weights [512,64] f32 -> [64,512] bf16 ----
__global__ void prep_weights(const float* __restrict__ Wq, const float* __restrict__ Wk,
                             const float* __restrict__ Wv,
                             __bf16* __restrict__ Wtq, __bf16* __restrict__ Wtk,
                             __bf16* __restrict__ Wtv) {
    int idx = blockIdx.x * blockDim.x + threadIdx.x;
    if (idx >= D_MODEL * DH) return;
    int k = idx / DH, c = idx % DH;
    int t = c * D_MODEL + k;
    Wtq[t] = (__bf16)Wq[idx];
    Wtk[t] = (__bf16)Wk[idx];
    Wtv[t] = (__bf16)Wv[idx];
}

// ---- Kernel 1: QKV projection, one wave per 16-row tile ----
// Q is pre-scaled by 1/sqrt(64); V is stored transposed as Vt[b][d][n].
__global__ void __launch_bounds__(32)
qkv_proj(const float* __restrict__ x,
         const __bf16* __restrict__ Wtq, const __bf16* __restrict__ Wtk,
         const __bf16* __restrict__ Wtv,
         const float* __restrict__ bq, const float* __restrict__ bk,
         const float* __restrict__ bv,
         __bf16* __restrict__ Qb, __bf16* __restrict__ Kb, __bf16* __restrict__ Vt) {
    const int lane = threadIdx.x & 31;
    const int n16  = lane & 15;
    const int h    = lane >> 4;
    const int row0 = blockIdx.x * 16;          // global row in [0, B*N)

    v8f cq[4] = {}, ck[4] = {}, cv[4] = {};

    for (int k0 = 0; k0 < D_MODEL; k0 += 32) {
        v16bf a = load_a_f32(x + (size_t)(row0 + n16) * D_MODEL + k0, h);
#pragma unroll
        for (int t = 0; t < 4; ++t) {
            const int wrow = (t * 16 + n16) * D_MODEL + k0;  // Wt row = W column
            cq[t] = wmma_bf16(a, load_b_bf16(Wtq + wrow, h), cq[t]);
            ck[t] = wmma_bf16(a, load_b_bf16(Wtk + wrow, h), ck[t]);
            cv[t] = wmma_bf16(a, load_b_bf16(Wtv + wrow, h), cv[t]);
        }
    }

    const int batch = row0 / SEQ;
#pragma unroll
    for (int t = 0; t < 4; ++t) {
        const int col = t * 16 + n16;
        const float bqv = bq[col], bkv = bk[col], bvv = bv[col];
#pragma unroll
        for (int j = 0; j < 8; ++j) {
            const int rg = row0 + j + h * 8;   // C layout: row = j + h*8
            Qb[(size_t)rg * DH + col] = (__bf16)((cq[t][j] + bqv) * 0.125f);
            Kb[(size_t)rg * DH + col] = (__bf16)(ck[t][j] + bkv);
            const int nloc = rg - batch * SEQ;
            Vt[(size_t)batch * DH * SEQ + (size_t)col * SEQ + nloc] =
                (__bf16)(cv[t][j] + bvv);
        }
    }
}

// ---- Kernel 2: flash attention ----
// 4 waves / block; each wave owns 32 Q rows (2 row-tiles); block = 128 Q rows.
// K/V kv-tiles double-buffered in LDS via async global->LDS copies: stage
// chunk i+1, compute chunk i, gate on ASYNCcnt + barrier.
#define KPAD 72   // 32x64 K tile row stride (bf16): 144B, 16B aligned, bank-skewed
#define VPAD 40   // 64x32 V^T tile row stride (bf16): 80B, 16B aligned

__global__ void __launch_bounds__(128)
flash_attn(const __bf16* __restrict__ Qb, const __bf16* __restrict__ Kb,
           const __bf16* __restrict__ Vt, float* __restrict__ out) {
    __shared__ __align__(16) __bf16 Ks[2][32 * KPAD];   // K rows (kv-major)
    __shared__ __align__(16) __bf16 Vs[2][64 * VPAD];   // V^T rows (head-dim-major)
    __shared__ __align__(16) __bf16 Pb[4][16 * 32];     // per-wave P repack

    const int tid   = threadIdx.x;
    const int wave  = tid >> 5;
    const int lane  = tid & 31;
    const int n16   = lane & 15;
    const int h     = lane >> 4;
    const int qblk  = blockIdx.x * 128;
    const int q0    = qblk + wave * 32;                 // wave's 2 row-tiles
    const int batch = qblk / SEQ;                       // 128 | 4096
    const __bf16* Kbase = Kb + (size_t)batch * SEQ * DH;
    const __bf16* Vbase = Vt + (size_t)batch * DH * SEQ;

    // Per-thread staging assignment: 2 K b128-chunks + 2 V b128-chunks.
    const int kr0 = tid >> 3,         kc0 = (tid & 7) * 8;       // K rows 0..15
    const int kr1 = (tid + 128) >> 3, kc1 = kc0;                 // K rows 16..31
    const int vr0 = tid >> 2,         vc0 = (tid & 3) * 8;       // V rows 0..31
    const int vr1 = (tid + 128) >> 2, vc1 = vc0;                 // V rows 32..63

    // Q fragments: 2 row-tiles x 2 k-steps, resident for whole kernel.
    v16bf aq[2][2];
#pragma unroll
    for (int rt = 0; rt < 2; ++rt) {
        const __bf16* qrow = Qb + (size_t)(q0 + rt * 16 + n16) * DH;
        aq[rt][0] = load_a_bf16(qrow, h);
        aq[rt][1] = load_a_bf16(qrow + 32, h);
    }

    v8f o[2][4] = {};
    float mrow[2][8], lrow[2][8];
#pragma unroll
    for (int rt = 0; rt < 2; ++rt)
#pragma unroll
        for (int j = 0; j < 8; ++j) { mrow[rt][j] = -1e30f; lrow[rt][j] = 0.0f; }

    // Prologue: stage chunk 0 into buffer 0.
    {
        copy_b128_g2l(&Ks[0][kr0 * KPAD + kc0], Kbase + (size_t)kr0 * DH + kc0);
        copy_b128_g2l(&Ks[0][kr1 * KPAD + kc1], Kbase + (size_t)kr1 * DH + kc1);
        copy_b128_g2l(&Vs[0][vr0 * VPAD + vc0], Vbase + (size_t)vr0 * SEQ + vc0);
        copy_b128_g2l(&Vs[0][vr1 * VPAD + vc1], Vbase + (size_t)vr1 * SEQ + vc1);
    }

    for (int kv = 0, it = 0; kv < SEQ; kv += 32, ++it) {
        const int cur = it & 1;
        wait_async0();          // this wave's staged loads have landed in LDS
        __syncthreads();        // all waves' loads landed; prev chunk consumed

        // Stage next chunk into the other buffer while we compute this one.
        if (kv + 32 < SEQ) {
            const int nxt = cur ^ 1;
            const int kn  = kv + 32;
            copy_b128_g2l(&Ks[nxt][kr0 * KPAD + kc0], Kbase + (size_t)(kn + kr0) * DH + kc0);
            copy_b128_g2l(&Ks[nxt][kr1 * KPAD + kc1], Kbase + (size_t)(kn + kr1) * DH + kc1);
            copy_b128_g2l(&Vs[nxt][vr0 * VPAD + vc0], Vbase + (size_t)vr0 * SEQ + kn + vc0);
            copy_b128_g2l(&Vs[nxt][vr1 * VPAD + vc1], Vbase + (size_t)vr1 * SEQ + kn + vc1);
        }

#pragma unroll
        for (int rt = 0; rt < 2; ++rt) {
            // ---- S = (Q/sqrt(d)) @ K^T : two 16x16 tiles, B-frags from LDS ----
            const __bf16* kp0 = &Ks[cur][(size_t)n16 * KPAD];
            const __bf16* kp1 = &Ks[cur][(size_t)(16 + n16) * KPAD];
            v8f s0 = {}, s1 = {};
            s0 = wmma_bf16(aq[rt][0], load_b_bf16(kp0, h), s0);
            s0 = wmma_bf16(aq[rt][1], load_b_bf16(kp0 + 32, h), s0);
            s1 = wmma_bf16(aq[rt][0], load_b_bf16(kp1, h), s1);
            s1 = wmma_bf16(aq[rt][1], load_b_bf16(kp1 + 32, h), s1);

            // ---- online softmax; row j+h*8 spans one 16-lane half ----
            float p0[8], p1[8];
#pragma unroll
            for (int j = 0; j < 8; ++j) {
                float vm = fmaxf(s0[j], s1[j]);
                vm = fmaxf(vm, __shfl_xor(vm, 1, 16));
                vm = fmaxf(vm, __shfl_xor(vm, 2, 16));
                vm = fmaxf(vm, __shfl_xor(vm, 4, 16));
                vm = fmaxf(vm, __shfl_xor(vm, 8, 16));
                const float mnew  = fmaxf(mrow[rt][j], vm);
                const float alpha = __expf(mrow[rt][j] - mnew);
                mrow[rt][j] = mnew;
                p0[j] = __expf(s0[j] - mnew);
                p1[j] = __expf(s1[j] - mnew);
                float rs = p0[j] + p1[j];
                rs += __shfl_xor(rs, 1, 16);
                rs += __shfl_xor(rs, 2, 16);
                rs += __shfl_xor(rs, 4, 16);
                rs += __shfl_xor(rs, 8, 16);
                lrow[rt][j] = lrow[rt][j] * alpha + rs;
#pragma unroll
                for (int t = 0; t < 4; ++t) o[rt][t][j] *= alpha;
            }

            // ---- repack P (C layout -> A layout) via this wave's LDS buffer ----
            __bf16* pb = &Pb[wave][0];
#pragma unroll
            for (int j = 0; j < 8; ++j) {
                const int r = j + h * 8;
                pb[r * 32 + n16]      = (__bf16)p0[j];
                pb[r * 32 + 16 + n16] = (__bf16)p1[j];
            }
            // private buffer, single wave: in-order LDS + explicit DScnt wait
            asm volatile("s_wait_dscnt 0" ::: "memory");
            const v16bf ap = load_a_bf16(pb + n16 * 32, h);

            // ---- O += P @ V : B columns = V^T rows in LDS ----
#pragma unroll
            for (int t = 0; t < 4; ++t)
                o[rt][t] = wmma_bf16(
                    ap, load_b_bf16(&Vs[cur][(size_t)(t * 16 + n16) * VPAD], h),
                    o[rt][t]);
        }
    }

    // ---- epilogue: normalize and store fp32 ----
#pragma unroll
    for (int rt = 0; rt < 2; ++rt)
#pragma unroll
        for (int t = 0; t < 4; ++t)
#pragma unroll
            for (int j = 0; j < 8; ++j) {
                const int rg = q0 + rt * 16 + j + h * 8;
                out[(size_t)rg * DH + t * 16 + n16] = o[rt][t][j] / lrow[rt][j];
            }
}

extern "C" void kernel_launch(void* const* d_in, const int* in_sizes, int n_in,
                              void* d_out, int out_size, void* d_ws, size_t ws_size,
                              hipStream_t stream) {
    (void)in_sizes; (void)n_in; (void)out_size; (void)ws_size;
    const float* x  = (const float*)d_in[0];
    const float* e  = (const float*)d_in[1];
    const float* Wq = (const float*)d_in[2];
    const float* bq = (const float*)d_in[3];
    const float* Wk = (const float*)d_in[4];
    const float* bk = (const float*)d_in[5];
    const float* Wv = (const float*)d_in[6];
    const float* bv = (const float*)d_in[7];
    float* out = (float*)d_out;

    // Workspace layout (bytes): Wt_q/k/v bf16 64KB each, then Qb/Kb/Vt bf16 4MB each.
    char* ws = (char*)d_ws;
    __bf16* Wtq = (__bf16*)(ws);
    __bf16* Wtk = (__bf16*)(ws + (64u << 10));
    __bf16* Wtv = (__bf16*)(ws + (128u << 10));
    __bf16* Qb  = (__bf16*)(ws + (256u << 10));
    __bf16* Kb  = Qb + (size_t)BATCH * SEQ * DH;
    __bf16* Vt  = Kb + (size_t)BATCH * SEQ * DH;

    prep_weights<<<(D_MODEL * DH + 255) / 256, 256, 0, stream>>>(Wq, Wk, Wv, Wtq, Wtk, Wtv);
    qkv_proj<<<BATCH * SEQ / 16, 32, 0, stream>>>(x, Wtq, Wtk, Wtv, bq, bk, bv, Qb, Kb, Vt);
    flash_attn<<<BATCH * SEQ / 128, 128, 0, stream>>>(Qb, Kb, Vt, out);

    // Passthrough of e into the second half of the output tuple.
    (void)hipMemcpyAsync(out + (size_t)BATCH * SEQ * DH, e,
                         (size_t)BATCH * SEQ * DH * sizeof(float),
                         hipMemcpyDeviceToDevice, stream);
}
